// ElectronicEmbedding_50525995270234
// MI455X (gfx1250) — matrix operand order repaired and checked by reading, not compiled, not run on hardware
//
#include <hip/hip_runtime.h>

#define PH 136                  // LDS pitch in bf16 halves (128 + 8 pad)
#define EPSC 1e-8f
#define LOG2C 0.69314718055994530942f

typedef __attribute__((ext_vector_type(16))) __bf16 v16bf;
typedef __attribute__((ext_vector_type(8)))  __bf16 v8bf;
typedef __attribute__((ext_vector_type(8)))  float  v8f;

// shifted softplus: softplus(x) - log(2), branch-free fast form
//   softplus(x) = max(x,0) + log(1 + exp(-|x|))
__device__ __forceinline__ float sspf(float x) {
    float t = __expf(-fabsf(x));
    return fmaxf(x, 0.f) + __logf(1.0f + t) - LOG2C;
}
__device__ __forceinline__ unsigned short f2bf(float f) {
    unsigned u = __float_as_uint(f);
    u = (u + 0x7FFFu + ((u >> 16) & 1u)) >> 16;   // RNE
    return (unsigned short)u;
}
__device__ __forceinline__ unsigned f2ord(float f) {
    unsigned u = __float_as_uint(f);
    return (u & 0x80000000u) ? ~u : (u | 0x80000000u);
}
__device__ __forceinline__ float ord2f(unsigned u) {
    unsigned b = (u & 0x80000000u) ? (u & 0x7FFFFFFFu) : ~u;
    return __uint_as_float(b);
}

// ---------------------------------------------------------------- small prep
__global__ void ee_init(float* T, float* S, unsigned* Mmax, int B) {
    int i = blockIdx.x * blockDim.x + threadIdx.x;
    if (i < B) T[i] = 0.f;
    if (i == 0) { *S = 0.f; *Mmax = 0u; }
}

__global__ void ee_mol(const float* __restrict__ ef, float* e0, float* e1,
                       float* e0n, float* e1n, int B) {
    int i = blockIdx.x * blockDim.x + threadIdx.x;
    if (i < B) {
        float f = ef[i];
        float p = fmaxf(f, 0.f), n = fmaxf(-f, 0.f);
        e0[i] = p; e1[i] = n;
        e0n[i] = p / fmaxf(p, 1.f);
        e1n[i] = n / fmaxf(n, 1.f);
    }
}

// A0 = Wk[:,0] @ Wq, A1 = Wk[:,1] @ Wq, c = Wk[:,c] . bq   (1 block x 128)
__global__ void ee_rank(const float* __restrict__ Wq, const float* __restrict__ bq,
                        const float* __restrict__ Wk, float* A0, float* A1, float* c01) {
    int t = threadIdx.x;
    float a0 = 0.f, a1 = 0.f;
    for (int j = 0; j < 128; ++j) {
        float wq = Wq[j * 128 + t];
        a0 += Wk[2 * j] * wq;
        a1 += Wk[2 * j + 1] * wq;
    }
    A0[t] = a0; A1[t] = a1;
    __shared__ float p0[128], p1[128];
    p0[t] = Wk[2 * t] * bq[t];
    p1[t] = Wk[2 * t + 1] * bq[t];
    __syncthreads();
    for (int s = 64; s > 0; s >>= 1) {
        if (t < s) { p0[t] += p0[t + s]; p1[t] += p1[t + s]; }
        __syncthreads();
    }
    if (t == 0) { c01[0] = p0[0]; c01[1] = p1[0]; }
}

// convert W1|W2|Wo to bf16 (3*16384 contiguous)
__global__ void ee_conv(const float* __restrict__ W1, const float* __restrict__ W2,
                        const float* __restrict__ Wo, unsigned short* __restrict__ Wh) {
    int i = blockIdx.x * blockDim.x + threadIdx.x;
    if (i < 16384)       Wh[i] = f2bf(W1[i]);
    else if (i < 32768)  Wh[i] = f2bf(W2[i - 16384]);
    else if (i < 49152)  Wh[i] = f2bf(Wo[i - 32768]);
}

// ------------------------------------------------- w_i + global max (1 wave/atom)
__global__ void ee_w(const float* __restrict__ X, const int* __restrict__ idx,
                     const float* __restrict__ A0g, const float* __restrict__ A1g,
                     const float* __restrict__ c01, const float* __restrict__ e0n,
                     const float* __restrict__ e1n, float* __restrict__ wbuf,
                     unsigned* Mmax, int N) {
    int lane = threadIdx.x & 31;
    int wv = threadIdx.x >> 5;
    int atom = blockIdx.x * 8 + wv;
    __shared__ float smax[8];
    float wval = -3.0e38f;
    if (atom < N) {
        float4 x  = ((const float4*)(X + (size_t)atom * 128))[lane];
        float4 a0 = ((const float4*)A0g)[lane];
        float4 a1 = ((const float4*)A1g)[lane];
        float d0 = x.x * a0.x + x.y * a0.y + x.z * a0.z + x.w * a0.w;
        float d1 = x.x * a1.x + x.y * a1.y + x.z * a1.z + x.w * a1.w;
        for (int s = 16; s > 0; s >>= 1) {
            d0 += __shfl_xor(d0, s, 32);
            d1 += __shfl_xor(d1, s, 32);
        }
        int m = idx[atom];
        wval = e0n[m] * (d0 + c01[0]) + e1n[m] * (d1 + c01[1]);
        if (lane == 0) wbuf[atom] = wval;
    }
    if (lane == 0) smax[wv] = wval;
    __syncthreads();
    if (threadIdx.x == 0) {
        float mx = smax[0];
        for (int i = 1; i < 8; ++i) mx = fmaxf(mx, smax[i]);
        atomicMax(Mmax, f2ord(mx));
    }
}

// ---------------------------------------- exp + segment sums + global sum
__global__ void ee_exp(float* __restrict__ wbuf, const int* __restrict__ idx,
                       const unsigned* __restrict__ Mmax, float* __restrict__ T,
                       float* __restrict__ S, int N) {
    int i = blockIdx.x * blockDim.x + threadIdx.x;
    float M = ord2f(*Mmax);
    float e = 0.f;
    if (i < N) {
        e = __expf(wbuf[i] - M);
        wbuf[i] = e;                      // in-place: w -> exp(w - M)
        atomicAdd(T + idx[i], e);
    }
    for (int s = 16; s > 0; s >>= 1) e += __shfl_xor(e, s, 32);
    __shared__ float ps[8];
    if ((threadIdx.x & 31) == 0) ps[threadIdx.x >> 5] = e;
    __syncthreads();
    if (threadIdx.x == 0) {
        float t = 0.f;
        for (int k = 0; k < 8; ++k) t += ps[k];
        atomicAdd(S, t);
    }
}

// ------------------------------------------------- WMMA fragment loaders (LDS)
// A (16x32 bf16): lane L -> row L&15; K = (L>>4)*8 + {0..7} and +16..+23
__device__ __forceinline__ v16bf load_afrag(const unsigned short* ab, int lane, int kc) {
    int row = lane & 15;
    int base = (lane >> 4) * 8;
    const unsigned short* p = ab + row * PH + kc * 32 + base;
    union { v16bf v; v8bf h[2]; } u;
    u.h[0] = *(const v8bf*)p;
    u.h[1] = *(const v8bf*)(p + 16);
    return u.v;
}
// B (32x16 bf16): lane L -> col L&15; K = (L>>4)*8 + {0..7} and +16..+23
__device__ __forceinline__ v16bf load_bfrag(const unsigned short* wl, int lane, int nt, int kc) {
    int n = lane & 15;
    int k0 = kc * 32 + (lane >> 4) * 8;
    const unsigned short* p = wl + (nt * 16 + n) * PH + k0;
    union { v16bf v; v8bf h[2]; } u;
    u.h[0] = *(const v8bf*)p;
    u.h[1] = *(const v8bf*)(p + 16);
    return u.v;
}

#define WMMA_BF16(Afr, Bfr, Cacc) \
    __builtin_amdgcn_wmma_f32_16x16x32_bf16(false, (Afr), false, (Bfr), (short)0, (Cacc), false, false)

// ------------------------------------- fused  x -> ssp -> W1 -> ssp -> W2 -> +x -> ssp -> Wo
__global__ void __launch_bounds__(256)
ee_mlp(const int* __restrict__ idx, const float* __restrict__ ew,
       const float* __restrict__ T, const float* __restrict__ Sp,
       const float* __restrict__ e0, const float* __restrict__ e1,
       const float* __restrict__ Wv, const unsigned short* __restrict__ Wh,
       float* __restrict__ out, int N) {
    extern __shared__ char smem[];
    unsigned short* Wl = (unsigned short*)smem;           // 3 * 128 * PH halves
    float* Wv0 = (float*)(smem + 3 * 128 * PH * 2);       // 128 f32
    float* Wv1 = Wv0 + 128;
    char* wavemem = (char*)(Wv1 + 128);

    int tid = threadIdx.x;
    int lane = tid & 31;
    int wv = tid >> 5;
    unsigned short* abA = (unsigned short*)(wavemem + wv * 8832);
    unsigned short* abB = abA + 16 * PH;
    float* srow = (float*)((char*)abA + 8704);            // (s0,s1) x 16 rows

    // stage bf16 weights with padded pitch (uint = 2 halves)
    {
        const unsigned* src = (const unsigned*)Wh;
        unsigned* dst = (unsigned*)Wl;
        for (int u = tid; u < 3 * 8192; u += 256) {
            int mat = u >> 13;
            int r = u & 8191;
            int row = r >> 6, cu = r & 63;
            dst[mat * (128 * (PH / 2)) + row * (PH / 2) + cu] = src[u];
        }
        if (tid < 128) { Wv0[tid] = Wv[2 * tid]; Wv1[tid] = Wv[2 * tid + 1]; }
    }
    __syncthreads();

    float epsS = EPSC * Sp[0];
    int numTiles = (N + 127) >> 7;
    const unsigned short* W1l = Wl;
    const unsigned short* W2l = Wl + 128 * PH;
    const unsigned short* Wol = Wl + 2 * 128 * PH;
    int colLane = lane & 15;
    int mHi = (lane >> 4) * 8;

    for (int tile = blockIdx.x; tile < numTiles; tile += gridDim.x) {
        int rowBase = tile * 128 + wv * 16;

        // ---- build act0 = ssp(x), x_i = s0*Wv0 + s1*Wv1 (rank-2 rows)
        for (int rr = 0; rr < 16; ++rr) {
            int row = rowBase + rr;
            float s0 = 0.f, s1 = 0.f;
            if (row < N) {
                int m = idx[row];
                float sc = ew[row] / (T[m] + epsS);
                s0 = sc * e0[m];
                s1 = sc * e1[m];
            }
            if (lane == 0) { srow[2 * rr] = s0; srow[2 * rr + 1] = s1; }
            int c0 = lane * 4;
            float x0 = s0 * Wv0[c0 + 0] + s1 * Wv1[c0 + 0];
            float x1 = s0 * Wv0[c0 + 1] + s1 * Wv1[c0 + 1];
            float x2 = s0 * Wv0[c0 + 2] + s1 * Wv1[c0 + 2];
            float x3 = s0 * Wv0[c0 + 3] + s1 * Wv1[c0 + 3];
            uint2 pv;
            pv.x = (unsigned)f2bf(sspf(x0)) | ((unsigned)f2bf(sspf(x1)) << 16);
            pv.y = (unsigned)f2bf(sspf(x2)) | ((unsigned)f2bf(sspf(x3)) << 16);
            *(uint2*)(abA + rr * PH + c0) = pv;
        }

        // ---- layer 1: act1 = ssp(act0 @ W1^T)  -> abB
        {
            v16bf Af0 = load_afrag(abA, lane, 0);
            v16bf Af1 = load_afrag(abA, lane, 1);
            v16bf Af2 = load_afrag(abA, lane, 2);
            v16bf Af3 = load_afrag(abA, lane, 3);
            for (int nt = 0; nt < 8; ++nt) {
                v8f acc = {0.f, 0.f, 0.f, 0.f, 0.f, 0.f, 0.f, 0.f};
                acc = WMMA_BF16(Af0, load_bfrag(W1l, lane, nt, 0), acc);
                acc = WMMA_BF16(Af1, load_bfrag(W1l, lane, nt, 1), acc);
                acc = WMMA_BF16(Af2, load_bfrag(W1l, lane, nt, 2), acc);
                acc = WMMA_BF16(Af3, load_bfrag(W1l, lane, nt, 3), acc);
                int col = nt * 16 + colLane;
#pragma unroll
                for (int r = 0; r < 8; ++r)
                    abB[(mHi + r) * PH + col] = f2bf(sspf(acc[r]));
            }
        }

        // ---- layer 2: y = x + act1 @ W2^T ; act2 = ssp(y) -> abA
        {
            v16bf Af0 = load_afrag(abB, lane, 0);
            v16bf Af1 = load_afrag(abB, lane, 1);
            v16bf Af2 = load_afrag(abB, lane, 2);
            v16bf Af3 = load_afrag(abB, lane, 3);
            for (int nt = 0; nt < 8; ++nt) {
                v8f acc = {0.f, 0.f, 0.f, 0.f, 0.f, 0.f, 0.f, 0.f};
                acc = WMMA_BF16(Af0, load_bfrag(W2l, lane, nt, 0), acc);
                acc = WMMA_BF16(Af1, load_bfrag(W2l, lane, nt, 1), acc);
                acc = WMMA_BF16(Af2, load_bfrag(W2l, lane, nt, 2), acc);
                acc = WMMA_BF16(Af3, load_bfrag(W2l, lane, nt, 3), acc);
                int col = nt * 16 + colLane;
                float wv0c = Wv0[col], wv1c = Wv1[col];
#pragma unroll
                for (int r = 0; r < 8; ++r) {
                    int M = mHi + r;
                    float xv = srow[2 * M] * wv0c + srow[2 * M + 1] * wv1c; // residual x
                    abA[M * PH + col] = f2bf(sspf(acc[r] + xv));
                }
            }
        }

        // ---- layer 3: out = act2 @ Wo^T -> global
        {
            v16bf Af0 = load_afrag(abA, lane, 0);
            v16bf Af1 = load_afrag(abA, lane, 1);
            v16bf Af2 = load_afrag(abA, lane, 2);
            v16bf Af3 = load_afrag(abA, lane, 3);
            for (int nt = 0; nt < 8; ++nt) {
                v8f acc = {0.f, 0.f, 0.f, 0.f, 0.f, 0.f, 0.f, 0.f};
                acc = WMMA_BF16(Af0, load_bfrag(Wol, lane, nt, 0), acc);
                acc = WMMA_BF16(Af1, load_bfrag(Wol, lane, nt, 1), acc);
                acc = WMMA_BF16(Af2, load_bfrag(Wol, lane, nt, 2), acc);
                acc = WMMA_BF16(Af3, load_bfrag(Wol, lane, nt, 3), acc);
                int col = nt * 16 + colLane;
#pragma unroll
                for (int r = 0; r < 8; ++r) {
                    int row = rowBase + mHi + r;
                    if (row < N) out[(size_t)row * 128 + col] = acc[r];
                }
            }
        }
    }
}

// ------------------------------------------------------------------ launcher
extern "C" void kernel_launch(void* const* d_in, const int* in_sizes, int n_in,
                              void* d_out, int out_size, void* d_ws, size_t ws_size,
                              hipStream_t stream) {
    const float* X   = (const float*)d_in[0];   // [N,128]
    const float* ef  = (const float*)d_in[1];   // [B]
    const int*   idx = (const int*)d_in[2];     // [N]
    const float* Wq  = (const float*)d_in[3];   // [128,128]
    const float* bq  = (const float*)d_in[4];   // [128]
    const float* Wk  = (const float*)d_in[5];   // [128,2]
    const float* Wv  = (const float*)d_in[6];   // [128,2]
    const float* W1  = (const float*)d_in[7];   // [128,128]
    const float* W2  = (const float*)d_in[8];   // [128,128]
    const float* Wo  = (const float*)d_in[9];   // [128,128]
    float* out = (float*)d_out;

    int N = in_sizes[0] / 128;
    int B = in_sizes[1];

    // workspace layout (floats)
    float* wsf = (float*)d_ws;
    float* A0  = wsf;                 // 128
    float* A1  = wsf + 128;           // 128
    float* c01 = wsf + 256;           // 2 (padded to 320)
    float* e0  = wsf + 320;           // B
    float* e1  = e0 + B;              // B
    float* e0n = e1 + B;              // B
    float* e1n = e0n + B;             // B
    float* T   = e1n + B;             // B
    float* S   = T + B;               // 1
    unsigned* Mmax = (unsigned*)(S + 1);
    float* wbuf = S + 8;              // N   (w, then exp(w-M))
    unsigned short* Wh = (unsigned short*)(wbuf + N);   // 3*16384 bf16

    ee_init<<<(B + 255) / 256, 256, 0, stream>>>(T, S, Mmax, B);
    ee_mol <<<(B + 255) / 256, 256, 0, stream>>>(ef, e0, e1, e0n, e1n, B);
    ee_rank<<<1, 128, 0, stream>>>(Wq, bq, Wk, A0, A1, c01);
    ee_conv<<<(49152 + 255) / 256, 256, 0, stream>>>(W1, W2, Wo, Wh);

    ee_w  <<<(N + 7) / 8, 256, 0, stream>>>(X, idx, A0, A1, c01, e0n, e1n, wbuf, Mmax, N);
    ee_exp<<<(N + 255) / 256, 256, 0, stream>>>(wbuf, idx, Mmax, T, S, N);

    const int SMEM = 3 * 128 * PH * 2 + 2 * 128 * 4 + 8 * 8832;   // 176128 B
    static_assert(3 * 128 * PH * 2 + 2 * 128 * 4 + 8 * 8832 == 176128, "lds layout");
    hipFuncSetAttribute(reinterpret_cast<const void*>(ee_mlp),
                        hipFuncAttributeMaxDynamicSharedMemorySize, SMEM);
    int numTiles = (N + 127) / 128;
    int grid = numTiles < 1024 ? numTiles : 1024;
    ee_mlp<<<grid, 256, SMEM, stream>>>(idx, wbuf, T, S, e0, e1, Wv, Wh, out, N);
}
